// MultiHeadAttention_7413113553430
// MI455X (gfx1250) — matrix operand order repaired and checked by reading, not compiled, not run on hardware
//
#include <hip/hip_runtime.h>

// MI455X / gfx1250 (CDNA5, wave32) fused RoPE multi-head attention.
// GEMMs: v_wmma_f32_16x16x32_f16, 8-wave 128x64 C-tiles, weights staged in LDS
// by the Tensor Data Mover (double-buffered, TENSORcnt-synced).
// Attention: transposed-logits flash attention (P lands in A-frag layout).

typedef __attribute__((ext_vector_type(16))) _Float16 v16h;
typedef __attribute__((ext_vector_type(8)))  _Float16 v8h;
typedef __attribute__((ext_vector_type(4)))  _Float16 v4h;
typedef __attribute__((ext_vector_type(8)))  float    v8f;
typedef __attribute__((ext_vector_type(4)))  float    v4f;
typedef __attribute__((ext_vector_type(4)))  int      v4i;
typedef __attribute__((ext_vector_type(8)))  int      i32x8;
typedef __attribute__((ext_vector_type(4)))  int      i32x4;
typedef __attribute__((ext_vector_type(4)))  unsigned u32x4;

namespace {
constexpr int kB  = 2;
constexpr int kS  = 2048;
constexpr int kD  = 1024;
constexpr int kH  = 16;
constexpr int kHD = 64;
constexpr int k3D = 3 * kD;
}

__device__ __forceinline__ v16h cat8(v8h lo, v8h hi) {
  return __builtin_shufflevector(lo, hi, 0,1,2,3,4,5,6,7,8,9,10,11,12,13,14,15);
}

__device__ __forceinline__ v8f wmma16x16x32(v16h a, v16h b, v8f c) {
  return __builtin_amdgcn_wmma_f32_16x16x32_f16(false, a, false, b, (short)0, c,
                                                false, false);
}

// ---------------------------------------------------------------- TDM helper
#if __has_include(<hip/amd_detail/amd_gfx1250_TDM.h>)
#define ATHENA_TDM_6ARG 1
#else
#define ATHENA_TDM_6ARG 0
#endif

__device__ __forceinline__ void tdm_load_2d_f16(
    unsigned lds_off_bytes, const void* gptr,
    unsigned tensor_d0, unsigned tensor_d1,
    unsigned tile_d0, unsigned tile_d1, unsigned stride0_elems) {
  const unsigned long long ga = (unsigned long long)(uintptr_t)gptr;
  u32x4 g0;
  g0[0] = 1u;                                          // count=1, user D#
  g0[1] = lds_off_bytes;                               // lds_addr [63:32]
  g0[2] = (unsigned)(ga & 0xffffffffu);                // global_addr[31:0]
  g0[3] = (unsigned)((ga >> 32) & 0x01ffffffu) | (2u << 30);  // addr[56:32]|type=2
  i32x8 g1;
  g1[0] = (int)(1u << 16);                             // data_size=1 (2 bytes)
  g1[1] = (int)((tensor_d0 & 0xffffu) << 16);          // tensor_dim0[15:0]
  g1[2] = (int)(((tensor_d0 >> 16) & 0xffffu) | ((tensor_d1 & 0xffffu) << 16));
  g1[3] = (int)(((tensor_d1 >> 16) & 0xffffu) | ((tile_d0 & 0xffffu) << 16));
  g1[4] = (int)(tile_d1 & 0xffffu);                    // tile_dim1 (tile_dim2=0)
  g1[5] = (int)stride0_elems;                          // dim0_stride[31:0]
  g1[6] = 0;
  g1[7] = 0;
  i32x4 gz = (i32x4)0;
#if ATHENA_TDM_6ARG
  i32x8 gz8 = (i32x8)0;
  __builtin_amdgcn_tensor_load_to_lds(g0, g1, gz, gz, gz8, 0);
#else
  __builtin_amdgcn_tensor_load_to_lds(g0, g1, gz, gz, 0);
#endif
}

// ---------------------------------------------------------------- fp32->fp16
__global__ __launch_bounds__(256) void cvt_f16_kernel(
    const float* __restrict__ src, _Float16* __restrict__ dst, int n4) {
  int i = blockIdx.x * blockDim.x + threadIdx.x;
  if (i < n4) {
    v4f s = *(const v4f*)(src + 4 * (size_t)i);
    v4h d;
    d[0] = (_Float16)s[0]; d[1] = (_Float16)s[1];
    d[2] = (_Float16)s[2]; d[3] = (_Float16)s[3];
    *(v4h*)(dst + 4 * (size_t)i) = d;
  }
}

// ------------------------------------------------------ GEMM: C = A @ B^T + b
// A: [M,K] f16 row-major. Bw: [N,K] f16 row-major. C: [M,N] f32.
// Block = 8 waves -> 128x64 C tile. B tile (64 rows x 64 K-halfs = 8KB) staged
// in LDS by the TDM, double-buffered; each transfer feeds 8 waves.
__global__ __launch_bounds__(256) void gemm_f16_wmma_bias(
    const _Float16* __restrict__ A, const _Float16* __restrict__ Bw,
    const float* __restrict__ bias, float* __restrict__ C,
    int M, int N, int K) {
  __shared__ _Float16 sB[2][64 * 64];     // 2 x 8KB double buffer

  const int lane = threadIdx.x & 31;
  const int w    = threadIdx.x >> 5;      // wave in block (0..7)
  const int tilesN = N >> 6;
  const int tn  = blockIdx.x % tilesN;
  const int tmb = blockIdx.x / tilesN;    // 128-row block index
  const int l15 = lane & 15;
  const int hi  = lane >> 4;
  const bool wave0 = (threadIdx.x < 32);

  const _Float16* arow = A + (size_t)(tmb * 128 + w * 16 + l15) * K;
  const _Float16* btile = Bw + (size_t)(tn * 64) * K;

  const int nch = K >> 6;                 // 64-wide K chunks

  if (wave0) {
    tdm_load_2d_f16((unsigned)(uintptr_t)&sB[0][0], btile,
                    (unsigned)K, (unsigned)N, 64u, 64u, (unsigned)K);
  }

  auto loadA = [&](int kc) -> v16h {
    v8h lo = *(const v8h*)(arow + kc + hi * 8);
    v8h hx = *(const v8h*)(arow + kc + 16 + hi * 8);
    return cat8(lo, hx);
  };

  v8f acc[4] = {};
  v16h aCur[2] = {loadA(0), loadA(32)};

  for (int i = 0; i < nch; ++i) {
    const int kc = i << 6;
    if (wave0) {
      const int kn = (i + 1 < nch) ? (kc + 64) : 0;   // wraps harmlessly
      tdm_load_2d_f16((unsigned)(uintptr_t)&sB[(i + 1) & 1][0],
                      (const char*)btile + (size_t)kn * 2,
                      (unsigned)K, (unsigned)N, 64u, 64u, (unsigned)K);
      __builtin_amdgcn_s_wait_tensorcnt((short)1);    // chunk i has landed
    }
    __syncthreads();

    // pipeline A for next chunk while consuming this one
    const int knA = (i + 1 < nch) ? (kc + 64) : kc;
    v16h aNext0 = loadA(knA);
    v16h aNext1 = loadA(knA + 32);
    __builtin_prefetch(arow + kc + 256, 0, 1);

    // load ALL B fragments of this chunk from LDS first, then run WMMA chain
    const _Float16* bb = &sB[i & 1][0];
    v16h bf[2][4];
#pragma unroll
    for (int ks = 0; ks < 2; ++ks)
#pragma unroll
      for (int t = 0; t < 4; ++t)
        bf[ks][t] =
            cat8(*(const v8h*)(bb + (t * 16 + l15) * 64 + ks * 32 + hi * 8),
                 *(const v8h*)(bb + (t * 16 + l15) * 64 + ks * 32 + 16 + hi * 8));
#pragma unroll
    for (int ks = 0; ks < 2; ++ks)
#pragma unroll
      for (int t = 0; t < 4; ++t)
        acc[t] = wmma16x16x32(aCur[ks], bf[ks][t], acc[t]);

    aCur[0] = aNext0;
    aCur[1] = aNext1;
    __syncthreads();   // all waves done with buf[i&1] before TDM refills it
  }

#pragma unroll
  for (int t = 0; t < 4; ++t) {
#pragma unroll
    for (int r = 0; r < 8; ++r) {
      const int rg = tmb * 128 + w * 16 + hi * 8 + r;
      const int cg = tn * 64 + t * 16 + l15;
      C[(size_t)rg * N + cg] = acc[t][r] + bias[cg];
    }
  }
}

// --------------------------------------- RoPE + head split + V transpose
__global__ __launch_bounds__(256) void rope_split_kernel(
    const float* __restrict__ qkv, const float* __restrict__ sp,
    _Float16* __restrict__ qh, _Float16* __restrict__ kh,
    _Float16* __restrict__ vt) {
  int i = blockIdx.x * blockDim.x + threadIdx.x;   // B*S*H*32 threads
  if (i >= kB * kS * kH * 32) return;
  const int d0 = i & 31;
  const int h  = (i >> 5) & 15;
  const int s  = (i >> 9) & (kS - 1);
  const int b  = i >> 20;

  const float* base = qkv + ((size_t)(b * kS + s)) * k3D + h * 192;
  const float sp0 = sp[s * kHD + d0];
  const float sp1 = sp[s * kHD + d0 + 32];
  const float c0 = __cosf(sp0), s0 = __sinf(sp0);
  const float c1 = __cosf(sp1), s1 = __sinf(sp1);

  const float q0 = base[d0],       q1 = base[d0 + 32];
  const float k0 = base[64 + d0],  k1 = base[96 + d0];
  const float v0 = base[128 + d0], v1 = base[160 + d0];

  const size_t row = ((size_t)(b * kH + h) * kS + s) * kHD;
  qh[row + d0]      = (_Float16)(q0 * c0 - q1 * s0);
  qh[row + d0 + 32] = (_Float16)(q1 * c1 + q0 * s1);
  kh[row + d0]      = (_Float16)(k0 * c0 - k1 * s0);
  kh[row + d0 + 32] = (_Float16)(k1 * c1 + k0 * s1);

  const size_t vbase = (size_t)(b * kH + h) * kHD * kS;
  vt[vbase + (size_t)d0 * kS + s]        = (_Float16)v0;
  vt[vbase + (size_t)(d0 + 32) * kS + s] = (_Float16)v1;
}

// ------------------------------------------------- flash attention per q-tile
__global__ __launch_bounds__(128) void attn_kernel(
    const _Float16* __restrict__ qh, const _Float16* __restrict__ kh,
    const _Float16* __restrict__ vt, const float* __restrict__ pos_bias,
    const int* __restrict__ msk, _Float16* __restrict__ valsh) {
  const int lane = threadIdx.x & 31;
  const int wid  = blockIdx.x * (blockDim.x >> 5) + (threadIdx.x >> 5);
  const int qt = wid & 127;
  const int h  = (wid >> 7) & 15;
  const int b  = wid >> 11;
  const int l15 = lane & 15;
  const int hi  = lane >> 4;

  const _Float16* qb = qh + (size_t)(b * kH + h) * kS * kHD;
  const _Float16* kb = kh + (size_t)(b * kH + h) * kS * kHD;
  const _Float16* vb = vt + (size_t)(b * kH + h) * kHD * kS;

  const int qrow = qt * 16 + l15;
  const v16h qf0 = cat8(*(const v8h*)(qb + (size_t)qrow * kHD + 0  + hi * 8),
                        *(const v8h*)(qb + (size_t)qrow * kHD + 16 + hi * 8));
  const v16h qf1 = cat8(*(const v8h*)(qb + (size_t)qrow * kHD + 32 + hi * 8),
                        *(const v8h*)(qb + (size_t)qrow * kHD + 48 + hi * 8));

  const float* pbrow = pos_bias + (size_t)h * kS * kS + (size_t)qrow * kS;
  const int*   mkrow = msk + (size_t)qrow * kS;

  float m_run = -3.0e38f;
  float l_run = 0.0f;
  v8f acc[4] = {};

  for (int n0 = 0; n0 < kS; n0 += 32) {
    // issue ALL loads for this step up front (K frags, V frags, pb, mask)
    v16h ka[2][2];
#pragma unroll
    for (int j = 0; j < 2; ++j) {
      const _Float16* krow = kb + (size_t)(n0 + j * 16 + l15) * kHD;
      ka[j][0] = cat8(*(const v8h*)(krow + hi * 8),
                      *(const v8h*)(krow + 16 + hi * 8));
      ka[j][1] = cat8(*(const v8h*)(krow + 32 + hi * 8),
                      *(const v8h*)(krow + 48 + hi * 8));
    }
    v16h vf[4];
#pragma unroll
    for (int t = 0; t < 4; ++t)
      vf[t] = cat8(*(const v8h*)(vb + (size_t)(t * 16 + l15) * kS + n0 + hi * 8),
                   *(const v8h*)(vb + (size_t)(t * 16 + l15) * kS + n0 + 16 + hi * 8));

    const int off0 = n0 + hi * 8;
    const v4f pb0a = *(const v4f*)(pbrow + off0);
    const v4f pb0b = *(const v4f*)(pbrow + off0 + 4);
    const v4f pb1a = *(const v4f*)(pbrow + off0 + 16);
    const v4f pb1b = *(const v4f*)(pbrow + off0 + 20);
    const v4i mk0a = *(const v4i*)(mkrow + off0);
    const v4i mk0b = *(const v4i*)(mkrow + off0 + 4);
    const v4i mk1a = *(const v4i*)(mkrow + off0 + 16);
    const v4i mk1b = *(const v4i*)(mkrow + off0 + 20);
    __builtin_prefetch(pbrow + n0 + 32 + hi * 8, 0, 1);
    __builtin_prefetch(mkrow + n0 + 32 + hi * 8, 0, 1);

    // S^T tiles: M=keys, N=qrows
    v8f st[2];
#pragma unroll
    for (int j = 0; j < 2; ++j) {
      v8f c = {};
      c = wmma16x16x32(ka[j][0], qf0, c);
      c = wmma16x16x32(ka[j][1], qf1, c);
      st[j] = c;
    }

    const float pbv[2][8] = {{pb0a[0], pb0a[1], pb0a[2], pb0a[3],
                              pb0b[0], pb0b[1], pb0b[2], pb0b[3]},
                             {pb1a[0], pb1a[1], pb1a[2], pb1a[3],
                              pb1b[0], pb1b[1], pb1b[2], pb1b[3]}};
    const int mkv[2][8] = {{mk0a[0], mk0a[1], mk0a[2], mk0a[3],
                            mk0b[0], mk0b[1], mk0b[2], mk0b[3]},
                           {mk1a[0], mk1a[1], mk1a[2], mk1a[3],
                            mk1b[0], mk1b[1], mk1b[2], mk1b[3]}};
    float p[2][8];
    float rowmax = -3.0e38f;
#pragma unroll
    for (int j = 0; j < 2; ++j)
#pragma unroll
      for (int r = 0; r < 8; ++r) {
        float lg = (st[j][r] + pbv[j][r]) * 0.125f;   // 1/sqrt(64)
        lg = (mkv[j][r] == 0) ? -9.0e15f : lg;
        p[j][r] = lg;
        rowmax = fmaxf(rowmax, lg);
      }
    rowmax = fmaxf(rowmax, __shfl_xor(rowmax, 16, 32));
    const float m_new = fmaxf(m_run, rowmax);
    const float corr  = __expf(m_run - m_new);
    float rs = 0.0f;
    v16h pf;
#pragma unroll
    for (int r = 0; r < 8; ++r) {
      const float e0 = __expf(p[0][r] - m_new);
      const float e1 = __expf(p[1][r] - m_new);
      rs += e0 + e1;
      pf[r]     = (_Float16)e0;
      pf[8 + r] = (_Float16)e1;
    }
    rs += __shfl_xor(rs, 16, 32);
    l_run = l_run * corr + rs;
    m_run = m_new;
#pragma unroll
    for (int t = 0; t < 4; ++t)
#pragma unroll
      for (int e = 0; e < 8; ++e) acc[t][e] *= corr;

#pragma unroll
    for (int t = 0; t < 4; ++t) acc[t] = wmma16x16x32(pf, vf[t], acc[t]);
  }

#pragma unroll
  for (int r = 0; r < 8; ++r) {
    const float ls  = __shfl(l_run, hi * 8 + r, 32);
    const float rcp = 1.0f / ls;
    const int rg = qt * 16 + hi * 8 + r;
#pragma unroll
    for (int t = 0; t < 4; ++t) {
      valsh[(size_t)(b * kS + rg) * kD + h * kHD + t * 16 + l15] =
          (_Float16)(acc[t][r] * rcp);
    }
  }
}

// --------------------------------------------------------------------- launch
extern "C" void kernel_launch(void* const* d_in, const int* in_sizes, int n_in,
                              void* d_out, int out_size, void* d_ws, size_t ws_size,
                              hipStream_t stream) {
  (void)in_sizes; (void)n_in; (void)out_size; (void)ws_size;
  const float* x        = (const float*)d_in[0];
  const float* pos_bias = (const float*)d_in[1];
  const float* sinpos   = (const float*)d_in[2];
  const int*   msk      = (const int*)d_in[3];
  const float* W_qkv    = (const float*)d_in[4];
  const float* b_qkv    = (const float*)d_in[5];
  const float* W_o      = (const float*)d_in[6];
  const float* b_o      = (const float*)d_in[7];
  float* out = (float*)d_out;

  char* ws = (char*)d_ws;
  size_t o = 0;
  _Float16* xh    = (_Float16*)(ws + o); o += (size_t)kB * kS * kD * 2;
  _Float16* wqkvh = (_Float16*)(ws + o); o += (size_t)k3D * kD * 2;
  _Float16* woh   = (_Float16*)(ws + o); o += (size_t)kD * kD * 2;
  float*    qkvf  = (float*)   (ws + o); o += (size_t)kB * kS * k3D * 4;
  _Float16* qh    = (_Float16*)(ws + o); o += (size_t)kB * kH * kS * kHD * 2;
  _Float16* kh    = (_Float16*)(ws + o); o += (size_t)kB * kH * kS * kHD * 2;
  _Float16* vt    = (_Float16*)(ws + o); o += (size_t)kB * kH * kHD * kS * 2;
  _Float16* valsh = (_Float16*)(ws + o); o += (size_t)kB * kS * kD * 2;

  const int M = kB * kS;   // 4096

  // 1) fp32 -> fp16 (x4 vectorized)
  cvt_f16_kernel<<<(M * kD / 4) / 256, 256, 0, stream>>>(x, xh, M * kD / 4);
  cvt_f16_kernel<<<(k3D * kD / 4) / 256, 256, 0, stream>>>(W_qkv, wqkvh, k3D * kD / 4);
  cvt_f16_kernel<<<(kD * kD / 4) / 256, 256, 0, stream>>>(W_o, woh, kD * kD / 4);

  // 2) qkv = x @ W_qkv^T + b_qkv   (M=4096, N=3072, K=1024)
  gemm_f16_wmma_bias<<<(M / 128) * (k3D / 64), 256, 0, stream>>>(
      xh, wqkvh, b_qkv, qkvf, M, k3D, kD);

  // 3) RoPE + split heads (+ V transpose)
  rope_split_kernel<<<(kB * kS * kH * 32) / 256, 256, 0, stream>>>(
      qkvf, sinpos, qh, kh, vt);

  // 4) flash attention: 4096 q-tiles, one wave each
  attn_kernel<<<(kB * kH * (kS / 16)) / 4, 128, 0, stream>>>(
      qh, kh, vt, pos_bias, msk, valsh);

  // 5) out = vals @ W_o^T + b_o    (M=4096, N=1024, K=1024)
  gemm_f16_wmma_bias<<<(M / 128) * (kD / 64), 256, 0, stream>>>(
      valsh, woh, b_o, out, M, kD, kD);
}
